// DirectTemporalNeRF_82660940578829
// MI455X (gfx1250) — matrix operand order repaired and checked by reading, not compiled, not run on hardware
//
#include <hip/hip_runtime.h>

typedef __attribute__((ext_vector_type(16))) _Float16 v16h;
typedef __attribute__((ext_vector_type(8)))  _Float16 v8h;
typedef __attribute__((ext_vector_type(8)))  float    v8f;
typedef __attribute__((ext_vector_type(4)))  int      v4i;

#define NPTS   262144
#define LVLS   12
#define TSIZE  131072
#define WIDTH  256
#define MBLK   64
#define KPAD   288
#define NLAYER 8

#define AS1 __attribute__((address_space(1)))
#define AS3 __attribute__((address_space(3)))

#if __has_builtin(__builtin_amdgcn_global_load_async_to_lds_b128)
#define HAVE_ASYNC_LDS 1
#else
#define HAVE_ASYNC_LDS 0
#endif

__device__ __forceinline__ void wait_async0() {
#if __has_builtin(__builtin_amdgcn_s_wait_asynccnt)
    __builtin_amdgcn_s_wait_asynccnt(0);
#else
    asm volatile("s_wait_asynccnt 0" ::: "memory");
#endif
}

// floor(16 * (256/16)^(i/11))
__device__ __constant__ float c_res[LVLS] = {
    16.f, 20.f, 26.f, 34.f, 43.f, 56.f, 72.f, 93.f, 120.f, 154.f, 198.f, 256.f};

// ---------------------------------------------------------------------------
// fp32 -> f16 weight conversion INTO WMMA B-fragment-swizzled layout.
// src: [K][256] row-major fp32.
// dst layout (halfs), per 32-row K panel kp:
//   off = kp*8192 + (ntile*32 + lane)*16 + h
//   where for element (k = kp*32+klo, n): ntile=n>>4, lane=(klo>>4)*16|(n&15),
//   h = klo&15.  Rows k >= K zero-filled.  A B fragment is then 16 contiguous
//   halfs per lane -> two ds_load_b128 in the GEMM kernel.
// ---------------------------------------------------------------------------
__global__ void cvt_w_swz(const float* __restrict__ src, _Float16* __restrict__ dst,
                          int K, int Kpad) {
    int idx = blockIdx.x * 256 + threadIdx.x;   // enumerates (k, n)
    if (idx >= Kpad * WIDTH) return;
    int k = idx / WIDTH, n = idx - k * WIDTH;
    float v = (k < K) ? src[idx] : 0.0f;
    int kp = k >> 5, klo = k & 31;
    int lane = ((klo >> 4) << 4) | (n & 15);
    int h = klo & 15;
    int ntile = n >> 4;
    dst[kp * 8192 + (ntile * 32 + lane) * 16 + h] = (_Float16)v;
}

// ---------------------------------------------------------------------------
// Fused width-256 MLP with skip concat after layer 4, WMMA f16 -> f32.
// MODE 0: time net   (input [pts(3), t],      head 256->3 -> dx + canonical)
// MODE 1: occ net    (input [canonical(3), blended(2)], head 256->4 -> out)
// ---------------------------------------------------------------------------
struct MlpParams {
    const _Float16* w;        // swizzled f16 weights, all 8 layers
    const float*    bias[8];  // per-layer bias [256]
    const float*    head_w;   // [256][HEADN]
    const float*    head_b;   // [HEADN]
    const float*    in_a;     // time: x [N,6]        occ: canonical [N,3]
    const float*    in_b;     // time: ts [N]         occ: blended [N,2]
    float*          out_main; // time: dx [N,3]       occ: out [N,4]
    float*          out_aux;  // time: canonical      occ: unused
};

template <int MODE>
__global__ __launch_bounds__(256) void mlp_fused(MlpParams P) {
    __shared__ _Float16 act[MBLK * KPAD];      // 36 KB  (row-major [m][k])
    __shared__ _Float16 wpanel[32 * WIDTH];    // 16 KB  (B-fragment swizzled)
    __shared__ float    skipf[MBLK * 8];       //  2 KB

    const int tid  = threadIdx.x;
    const int lane = tid & 31;
    const int wav  = tid >> 5;
    const int p0   = blockIdx.x * MBLK;

    constexpr int SKIPD = (MODE == 0) ? 3 : 5;  // skip-feat width
    constexpr int INITD = (MODE == 0) ? 4 : 5;  // layer-0 input width
    constexpr int HEADN = (MODE == 0) ? 3 : 4;

    // ---- init activations (zero-padded to KPAD) ----
    for (int idx = tid; idx < MBLK * KPAD; idx += 256) {
        int m = idx / KPAD, c = idx - m * KPAD;
        float v = 0.0f;
        if (c < INITD) {
            int p = p0 + m;
            if (MODE == 0) v = (c < 3) ? P.in_a[p * 6 + c] : P.in_b[p];
            else           v = (c < 3) ? P.in_a[p * 3 + c] : P.in_b[p * 2 + (c - 3)];
        }
        act[idx] = (_Float16)v;
    }
    for (int idx = tid; idx < MBLK * SKIPD; idx += 256) {
        int m = idx / SKIPD, c = idx - m * SKIPD;
        int p = p0 + m;
        float v = (MODE == 0) ? P.in_a[p * 6 + c]
                              : (c < 3 ? P.in_a[p * 3 + c] : P.in_b[p * 2 + (c - 3)]);
        skipf[m * 8 + c] = v;
    }
    __syncthreads();

    // K-panel counts and f16 weight offsets (in elements) per layer
    const int kp_tab[8]   = {1, 8, 8, 8, 8, 9, 8, 8};
    const int woff_tab[8] = {0, 8192, 73728, 139264, 204800, 270336, 344064, 409600};

    const int mt   = wav & 3;           // wave's M tile (0..3)
    const int nb   = (wav >> 2) * 8;    // wave's N-tile base (0 or 8)
    const int mrow = mt * 16 + (lane & 15);
    const int khi  = lane >> 4;
    const int ncol = (lane & 15);

    for (int l = 0; l < NLAYER; ++l) {
        const int Kp = kp_tab[l];
        const _Float16* gw = P.w + woff_tab[l];

        v8f acc[8] = {};

        for (int kp = 0; kp < Kp; ++kp) {
            // stage swizzled weight panel (8192 halfs = 16 KB) into LDS
            const _Float16* gp = gw + kp * 32 * WIDTH;
#if HAVE_ASYNC_LDS
#pragma unroll
            for (int i = 0; i < 4; ++i) {
                int e = tid * 8 + i * 2048;
                __builtin_amdgcn_global_load_async_to_lds_b128(
                    (AS1 v4i*)(gp + e), (AS3 v4i*)(&wpanel[e]), 0, 0);
            }
            wait_async0();
#else
#pragma unroll
            for (int i = 0; i < 4; ++i) {
                int e = tid * 8 + i * 2048;
                *(v8h*)&wpanel[e] = *(const v8h*)&gp[e];
            }
#endif
            if (kp + 1 < Kp)
                __builtin_prefetch(gw + (kp + 1) * 32 * WIDTH, 0, 0);
            __syncthreads();

            // A fragment (16x32 f16): lane m = L&15, K = {khi*16+0..15}
            union { v16h v; v8h h[2]; } af;
            const _Float16* arow = &act[mrow * KPAD + kp * 32 + khi * 16];
            af.h[0] = *(const v8h*)(arow);
            af.h[1] = *(const v8h*)(arow + 8);

#pragma unroll
            for (int t = 0; t < 8; ++t) {
                // B fragment: 16 contiguous halfs per lane in swizzled panel
                union { v16h v; v8h h[2]; } bf;
                const _Float16* brow = &wpanel[((nb + t) * 32 + lane) * 16];
                bf.h[0] = *(const v8h*)(brow);
                bf.h[1] = *(const v8h*)(brow + 8);
                acc[t] = __builtin_amdgcn_wmma_f32_16x16x32_f16(
                    false, af.v, false, bf.v, (short)0, acc[t], false, false);
            }
            __syncthreads();
        }

        // ---- bias + ReLU, write next activations ----
        const int cb = (l == 4) ? SKIPD : 0;  // skip concat after layer 4
        const float* bias = P.bias[l];
#pragma unroll
        for (int t = 0; t < 8; ++t) {
            const int n  = (nb + t) * 16 + ncol;
            const float bn = bias[n];
#pragma unroll
            for (int r = 0; r < 8; ++r) {
                const int m = mt * 16 + r + 8 * khi;   // C/D layout: M = r + 8*(L>>4)
                float v = acc[t][r] + bn;
                v = v > 0.0f ? v : 0.0f;
                act[m * KPAD + cb + n] = (_Float16)v;
            }
        }
        // zero pad columns [256+cb .. KPAD)
        {
            const int z0 = WIDTH + cb, padw = KPAD - z0;
            for (int idx = tid; idx < MBLK * padw; idx += 256) {
                int m = idx / padw, c = z0 + (idx - m * padw);
                act[m * KPAD + c] = (_Float16)0.0f;
            }
        }
        // skip features into columns [0 .. SKIPD)
        if (cb > 0) {
            for (int idx = tid; idx < MBLK * SKIPD; idx += 256) {
                int m = idx / SKIPD, c = idx - m * SKIPD;
                act[m * KPAD + c] = (_Float16)skipf[m * 8 + c];
            }
        }
        __syncthreads();
    }

    // ---- output head (tiny): 256 -> HEADN, fp32 accumulate ----
    const int tasks = MBLK * HEADN;
    if (tid < tasks) {
        int m = tid / HEADN, j = tid - m * HEADN;
        float s = P.head_b[j];
        for (int k = 0; k < WIDTH; ++k)
            s += (float)act[m * KPAD + k] * P.head_w[k * HEADN + j];
        int p = p0 + m;
        if (MODE == 0) {
            P.out_main[p * 3 + j] = s;                              // dx
            P.out_aux[p * 3 + j]  = (skipf[m * 8 + j] + s) / 1.5f;  // canonical
        } else {
            P.out_main[p * 4 + j] = s;                              // final out
        }
    }
}

// ---------------------------------------------------------------------------
// Instant-NGP hash encode + softmax blend over 12 levels -> blended [N,2]
// ---------------------------------------------------------------------------
__global__ __launch_bounds__(256) void hash_blend(
    const float* __restrict__ canonical, const float* __restrict__ ts,
    const float* __restrict__ tables, const float* __restrict__ bw,
    const float* __restrict__ bb, float* __restrict__ blended) {
    int p = blockIdx.x * 256 + threadIdx.x;
    if (p >= NPTS) return;
    float cx = canonical[p * 3 + 0];
    float cy = canonical[p * 3 + 1];
    float cz = canonical[p * 3 + 2];
    float t  = ts[p];

    // softmax(t * blend_w + blend_b)
    float lg[LVLS];
    float mx = -1e30f;
    for (int l = 0; l < LVLS; ++l) {
        lg[l] = t * bw[l] + bb[l];
        mx = fmaxf(mx, lg[l]);
    }
    float se = 0.0f;
    for (int l = 0; l < LVLS; ++l) { lg[l] = __expf(lg[l] - mx); se += lg[l]; }
    float inv = 1.0f / se;

    float b0 = 0.0f, b1 = 0.0f;
    for (int l = 0; l < LVLS; ++l) {
        float r  = c_res[l];
        float sx = cx * r, sy = cy * r, sz = cz * r;
        float fx = floorf(sx), fy = floorf(sy), fz = floorf(sz);
        float frx = sx - fx, fry = sy - fy, frz = sz - fz;
        unsigned bx = (unsigned)(int)fx, by = (unsigned)(int)fy, bz = (unsigned)(int)fz;
        const float* tab = tables + (size_t)l * TSIZE * 2;
        float f0 = 0.0f, f1 = 0.0f;
#pragma unroll
        for (int c = 0; c < 8; ++c) {
            unsigned ox = (c >> 2) & 1u, oy = (c >> 1) & 1u, oz = c & 1u;
            unsigned h = (bx + ox) * 1u ^ (by + oy) * 2654435761u ^ (bz + oz) * 805459861u;
            unsigned idx = h & (unsigned)(TSIZE - 1);
            float w = (ox ? frx : 1.0f - frx) * (oy ? fry : 1.0f - fry) *
                      (oz ? frz : 1.0f - frz);
            f0 += w * tab[idx * 2 + 0];
            f1 += w * tab[idx * 2 + 1];
        }
        float wl = lg[l] * inv;
        b0 += wl * f0;
        b1 += wl * f1;
    }
    blended[p * 2 + 0] = b0;
    blended[p * 2 + 1] = b1;
}

// ---------------------------------------------------------------------------
// Host launch.  Input order (lists flattened element-by-element):
//  0:x 1:ts 2-9:time_ws 10-17:time_bs 18:time_out_w 19:time_out_b
//  20:hash_tables 21:blend_w 22:blend_b 23-30:occ_ws 31-38:occ_bs 39:out_w 40:out_b
// d_out = [out (N*4) | dx (N*3)] fp32.
// ---------------------------------------------------------------------------
extern "C" void kernel_launch(void* const* d_in, const int* in_sizes, int n_in,
                              void* d_out, int out_size, void* d_ws, size_t ws_size,
                              hipStream_t stream) {
    (void)in_sizes; (void)n_in; (void)out_size; (void)ws_size;

    const float* x          = (const float*)d_in[0];
    const float* ts         = (const float*)d_in[1];
    const float* time_ws_[8]; for (int i = 0; i < 8; ++i) time_ws_[i] = (const float*)d_in[2 + i];
    const float* time_bs_[8]; for (int i = 0; i < 8; ++i) time_bs_[i] = (const float*)d_in[10 + i];
    const float* time_out_w = (const float*)d_in[18];
    const float* time_out_b = (const float*)d_in[19];
    const float* tables     = (const float*)d_in[20];
    const float* blend_w    = (const float*)d_in[21];
    const float* blend_b    = (const float*)d_in[22];
    const float* occ_ws_[8];  for (int i = 0; i < 8; ++i) occ_ws_[i] = (const float*)d_in[23 + i];
    const float* occ_bs_[8];  for (int i = 0; i < 8; ++i) occ_bs_[i] = (const float*)d_in[31 + i];
    const float* out_w      = (const float*)d_in[39];
    const float* out_b      = (const float*)d_in[40];

    // workspace layout
    char*      ws       = (char*)d_ws;
    _Float16*  wt_time  = (_Float16*)(ws);                   // 950 KB
    _Float16*  wt_occ   = (_Float16*)(ws + (1u << 20));      // 950 KB
    float*     canon    = (float*)(ws + (2u << 20));         // N*3 fp32
    float*     blended  = (float*)(ws + (5u << 20));         // N*2 fp32

    const int Ktime[8] = {4, 256, 256, 256, 256, 259, 256, 256};
    const int Kocc[8]  = {5, 256, 256, 256, 256, 261, 256, 256};
    const int Kpad_[8] = {32, 256, 256, 256, 256, 288, 256, 256};
    const int woff[8]  = {0, 8192, 73728, 139264, 204800, 270336, 344064, 409600};

    for (int l = 0; l < 8; ++l) {
        int ne = Kpad_[l] * WIDTH;
        cvt_w_swz<<<(ne + 255) / 256, 256, 0, stream>>>(time_ws_[l], wt_time + woff[l],
                                                        Ktime[l], Kpad_[l]);
        cvt_w_swz<<<(ne + 255) / 256, 256, 0, stream>>>(occ_ws_[l], wt_occ + woff[l],
                                                        Kocc[l], Kpad_[l]);
    }

    float* out = (float*)d_out;               // [N,4]
    float* dx  = out + (size_t)NPTS * 4;      // [N,3]

    MlpParams pt;
    pt.w = wt_time;
    for (int i = 0; i < 8; ++i) pt.bias[i] = time_bs_[i];
    pt.head_w = time_out_w; pt.head_b = time_out_b;
    pt.in_a = x; pt.in_b = ts;
    pt.out_main = dx; pt.out_aux = canon;
    mlp_fused<0><<<NPTS / MBLK, 256, 0, stream>>>(pt);

    hash_blend<<<(NPTS + 255) / 256, 256, 0, stream>>>(canon, ts, tables,
                                                       blend_w, blend_b, blended);

    MlpParams po;
    po.w = wt_occ;
    for (int i = 0; i < 8; ++i) po.bias[i] = occ_bs_[i];
    po.head_w = out_w; po.head_b = out_b;
    po.in_a = canon; po.in_b = blended;
    po.out_main = out; po.out_aux = nullptr;
    mlp_fused<1><<<NPTS / MBLK, 256, 0, stream>>>(po);
}
